// Net2_80444737454964
// MI455X (gfx1250) — matrix operand order repaired and checked by reading, not compiled, not run on hardware
//
#include <hip/hip_runtime.h>

#define BB 64
#define SS 764
#define SP 768
#define DD 256
#define HH 8
#define CC 32
#define BS (BB*SS)      // 48896
#define RT (BS/16)      // 3056 row tiles
#define NKT (SP/32)     // 24 key steps of 32

typedef __attribute__((ext_vector_type(16))) _Float16 v16h;
typedef __attribute__((ext_vector_type(8)))  _Float16 v8h;
typedef __attribute__((ext_vector_type(8)))  float    v8f;

static __device__ __forceinline__ v8f wmma16(v16h a, v16h b, v8f c) {
  // D = A(16x32 f16) * B(32x16 f16) + C(16x16 f32)
  return __builtin_amdgcn_wmma_f32_16x16x32_f16(false, a, false, b, (short)0, c,
                                                false, false);
}

static __device__ __forceinline__ v16h cat16(v8h a, v8h b) {
  v16h r;
#pragma unroll
  for (int i = 0; i < 8; ++i) { r[i] = a[i]; r[i + 8] = b[i]; }
  return r;
}

// ---------------- converters ----------------
__global__ void k_cvt(const float* __restrict__ src, _Float16* __restrict__ dst,
                      int n) {
  int i = blockIdx.x * blockDim.x + threadIdx.x;
  if (i < n) dst[i] = (_Float16)src[i];
}

// dst[n][k] = (f16) src[k][n] for 256x256 — makes WMMA B-fragments contiguous
__global__ void k_cvt_tr256(const float* __restrict__ src,
                            _Float16* __restrict__ dst) {
  int i = blockIdx.x * blockDim.x + threadIdx.x;  // 65536 threads
  int r = i >> 8, c = i & 255;
  dst[i] = (_Float16)src[c * 256 + r];
}

// zero the 4 padded rows (s = 764..767) of Q, K and Vt
__global__ void k_zero_pads(_Float16* __restrict__ Qb, _Float16* __restrict__ Kb,
                            _Float16* __restrict__ Vt) {
  int i = blockIdx.x * blockDim.x + threadIdx.x;  // 65536 = 512*4*32
  int c = i & 31, sp = (i >> 5) & 3, bh = i >> 7;
  int s = SS + sp;
  Qb[((size_t)bh * SP + s) * CC + c] = (_Float16)0.f;
  Kb[((size_t)bh * SP + s) * CC + c] = (_Float16)0.f;
  Vt[((size_t)bh * CC + c) * SP + s] = (_Float16)0.f;
}

// ---------------- fused QKV + gate projection ----------------
// block = 128 threads = 4 waves; wave w computes output w of {Q,K,V,gate}.
// Each wave register-blocks a 16x64 output strip (4 WMMA tiles, A reused 4x).
// grid = (3056 row tiles, 4 col strips of 64)
__global__ void __launch_bounds__(128) k_proj(
    const _Float16* __restrict__ Xq, const _Float16* __restrict__ Xm,
    const _Float16* __restrict__ Wq, const _Float16* __restrict__ Wk,
    const _Float16* __restrict__ Wv, const _Float16* __restrict__ Wg,
    const float* __restrict__ gb, _Float16* __restrict__ Qb,
    _Float16* __restrict__ Kb, _Float16* __restrict__ Vt,
    _Float16* __restrict__ gate) {
  const int wave = threadIdx.x >> 5;
  const int lane = threadIdx.x & 31;
  const int lo = lane & 15, hi = lane >> 4;
  const int row0 = blockIdx.x * 16;
  const int n0 = blockIdx.y * 64;
  const _Float16* X = (wave == 1 || wave == 2) ? Xm : Xq;
  const _Float16* W = (wave == 0) ? Wq : (wave == 1) ? Wk : (wave == 2) ? Wv : Wg;
  v8f acc[4] = {{}, {}, {}, {}};
  const _Float16* xrow = X + (size_t)(row0 + lo) * DD;
  const _Float16* wrow = W + (size_t)(n0 + lo) * DD;
#pragma unroll
  for (int kk = 0; kk < DD / 32; ++kk) {
    const int k0 = kk * 32;
    v16h a = cat16(*(const v8h*)(xrow + k0 + 16 * hi),
                   *(const v8h*)(xrow + k0 + 16 * hi + 8));
#pragma unroll
    for (int j = 0; j < 4; ++j) {
      const _Float16* wj = wrow + (size_t)j * 16 * DD + k0 + 16 * hi;
      v16h b = cat16(*(const v8h*)wj, *(const v8h*)(wj + 8));
      acc[j] = wmma16(a, b, acc[j]);
    }
  }
#pragma unroll
  for (int j = 0; j < 4; ++j) {
#pragma unroll
    for (int r = 0; r < 8; ++r) {
      int g = row0 + r + 8 * hi;          // global row in [0, B*S)
      int n = n0 + j * 16 + lo;           // output col = h*32 + c
      int h = n >> 5, c = n & 31;
      int b = g / SS, s = g - b * SS;
      float v = acc[j][r];
      if (wave == 0)
        Qb[((size_t)(b * HH + h) * SP + s) * CC + c] =
            (_Float16)(v * 0.17677669529663687f);  // * C^-0.5
      else if (wave == 1)
        Kb[((size_t)(b * HH + h) * SP + s) * CC + c] = (_Float16)v;
      else if (wave == 2)
        Vt[((size_t)(b * HH + h) * CC + c) * SP + s] = (_Float16)v;
      else {
        float x = v + gb[h * CC + c];
        gate[(size_t)g * DD + n] = (_Float16)(1.f / (1.f + __expf(-x)));
      }
    }
  }
}

// ---------------- flash attention per (b, h, 16-row q tile) ----------------
__global__ void __launch_bounds__(128) k_attn(
    const _Float16* __restrict__ Qb, const _Float16* __restrict__ Kb,
    const _Float16* __restrict__ Vt, const float* __restrict__ bias,
    const float* __restrict__ nb, const _Float16* __restrict__ gate,
    _Float16* __restrict__ G2) {
  __shared__ __align__(16) _Float16 ps[4][16][32];  // per-wave P staging
  const int wave = threadIdx.x >> 5;
  const int lane = threadIdx.x & 31;
  const int lo = lane & 15, hi = lane >> 4;
  const int job = blockIdx.x * 4 + wave;  // 24576 jobs
  const int qt = job % 48;
  const int h = (job / 48) & 7;
  const int b = job / 384;
  const int q0 = qt * 16;
  const _Float16* Qs = Qb + (size_t)(b * HH + h) * SP * CC;
  const _Float16* Ks = Kb + (size_t)(b * HH + h) * SP * CC;
  const _Float16* Vs = Vt + (size_t)(b * HH + h) * CC * SP;
  const float* biasb = bias + (size_t)b * SS;

  const _Float16* qrow = Qs + (size_t)(q0 + lo) * CC;
  v16h qf = cat16(*(const v8h*)(qrow + 16 * hi),
                  *(const v8h*)(qrow + 16 * hi + 8));  // K-dim = C = 32
  v16h onesf;
#pragma unroll
  for (int i = 0; i < 16; ++i) onesf[i] = (_Float16)1.f;

  float m[8];
  v8f o0 = {}, o1 = {}, lt = {};  // lt: running softmax denominator per row
#pragma unroll
  for (int r = 0; r < 8; ++r) m[r] = -1e30f;

  for (int kt = 0; kt < NKT; ++kt) {
    const int k0 = kt * 32;
    const _Float16* kr0 = Ks + (size_t)(k0 + lo) * CC + 16 * hi;
    const _Float16* kr1 = Ks + (size_t)(k0 + 16 + lo) * CC + 16 * hi;
    // speculative prefetch of next tile's K rows / V columns (global_prefetch_b8)
    __builtin_prefetch(kr0 + 32 * CC, 0, 3);
    __builtin_prefetch(kr1 + 32 * CC, 0, 3);
    v16h kf0 = cat16(*(const v8h*)kr0, *(const v8h*)(kr0 + 8));
    v16h kf1 = cat16(*(const v8h*)kr1, *(const v8h*)(kr1 + 8));
    v8f z = {};
    v8f s0 = wmma16(qf, kf0, z);   // logits cols k0..k0+15
    v8f s1 = wmma16(qf, kf1, z);   // logits cols k0+16..k0+31
    const int kA = k0 + lo, kB = k0 + 16 + lo;
    const bool mA = kA >= SS, mB = kB >= SS;
    const float biasA = biasb[mA ? (SS - 1) : kA];
    const float biasB = biasb[mB ? (SS - 1) : kB];
#pragma unroll
    for (int r = 0; r < 8; ++r) {
      int q = q0 + r + 8 * hi; if (q > SS - 1) q = SS - 1;
      const float* nrow = nb + ((size_t)h * SS + q) * SS;
      float a = mA ? -1e30f : (s0[r] + biasA + nrow[kA]);
      float c = mB ? -1e30f : (s1[r] + biasB + nrow[kB]);
      float rm = fmaxf(a, c);  // row max over the 16 lanes holding this row
      rm = fmaxf(rm, __shfl_xor(rm, 1, 32));
      rm = fmaxf(rm, __shfl_xor(rm, 2, 32));
      rm = fmaxf(rm, __shfl_xor(rm, 4, 32));
      rm = fmaxf(rm, __shfl_xor(rm, 8, 32));
      float mn = fmaxf(m[r], rm);
      float sc = __expf(m[r] - mn);
      float p0 = __expf(a - mn);
      float p1 = __expf(c - mn);
      m[r] = mn;
      o0[r] *= sc;
      o1[r] *= sc;
      lt[r] *= sc;                       // denominator rescales like O
      // C-layout -> LDS (re-fragmented below as A-layout)
      ps[wave][r + 8 * hi][lo] = (_Float16)p0;
      ps[wave][r + 8 * hi][16 + lo] = (_Float16)p1;
    }
    // P as A-fragment (16x32, K = key dim) — in-wave DS ordering guarantees RAW
    const _Float16* pr = &ps[wave][lo][0];
    v16h pf = cat16(*(const v8h*)(pr + 16 * hi), *(const v8h*)(pr + 16 * hi + 8));
    const _Float16* vr0 = Vs + (size_t)lo * SP + k0 + 16 * hi;
    const _Float16* vr1 = Vs + (size_t)(16 + lo) * SP + k0 + 16 * hi;
    __builtin_prefetch(vr0 + 32, 0, 3);
    __builtin_prefetch(vr1 + 32, 0, 3);
    v16h vf0 = cat16(*(const v8h*)vr0, *(const v8h*)(vr0 + 8));
    v16h vf1 = cat16(*(const v8h*)vr1, *(const v8h*)(vr1 + 8));
    o0 = wmma16(pf, vf0, o0);      // out cols c = 0..15
    o1 = wmma16(pf, vf1, o1);      // out cols c = 16..31
    lt = wmma16(pf, onesf, lt);    // row sums via matrix pipe (P x ones)
  }
#pragma unroll
  for (int r = 0; r < 8; ++r) {
    int q = q0 + r + 8 * hi;
    if (q < SS) {
      size_t g = (size_t)b * SS + q;
      float inv = 1.f / lt[r];
      size_t base = g * DD + h * CC;
      float g0 = (float)gate[base + lo];
      float g1 = (float)gate[base + 16 + lo];
      G2[base + lo] = (_Float16)(o0[r] * inv * g0);
      G2[base + 16 + lo] = (_Float16)(o1[r] * inv * g1);
    }
  }
}

// ---------------- output projection ----------------
// block = 128 = 4 waves, one 16-row tile per block; wave = 64-col strip.
__global__ void __launch_bounds__(128) k_outproj(
    const _Float16* __restrict__ G2, const _Float16* __restrict__ Wo,
    const float* __restrict__ ob, float* __restrict__ out) {
  const int wave = threadIdx.x >> 5;
  const int lane = threadIdx.x & 31;
  const int lo = lane & 15, hi = lane >> 4;
  const int row0 = blockIdx.x * 16;
  const int n0 = wave * 64;
  v8f acc[4] = {{}, {}, {}, {}};
  const _Float16* xrow = G2 + (size_t)(row0 + lo) * DD;
  const _Float16* wrow = Wo + (size_t)(n0 + lo) * DD;
#pragma unroll
  for (int kk = 0; kk < DD / 32; ++kk) {
    const int k0 = kk * 32;
    v16h a = cat16(*(const v8h*)(xrow + k0 + 16 * hi),
                   *(const v8h*)(xrow + k0 + 16 * hi + 8));
#pragma unroll
    for (int j = 0; j < 4; ++j) {
      const _Float16* wj = wrow + (size_t)j * 16 * DD + k0 + 16 * hi;
      v16h w = cat16(*(const v8h*)wj, *(const v8h*)(wj + 8));
      acc[j] = wmma16(a, w, acc[j]);
    }
  }
#pragma unroll
  for (int j = 0; j < 4; ++j) {
    const float bv = ob[n0 + j * 16 + lo];
#pragma unroll
    for (int r = 0; r < 8; ++r) {
      int g = row0 + r + 8 * hi;
      out[(size_t)g * DD + n0 + j * 16 + lo] = acc[j][r] + bv;
    }
  }
}

extern "C" void kernel_launch(void* const* d_in, const int* in_sizes, int n_in,
                              void* d_out, int out_size, void* d_ws,
                              size_t ws_size, hipStream_t stream) {
  (void)in_sizes; (void)n_in; (void)out_size; (void)ws_size;
  const float* q_data = (const float*)d_in[0];
  const float* m_data = (const float*)d_in[1];
  const float* bias   = (const float*)d_in[2];
  const float* nb     = (const float*)d_in[3];
  const float* q_w    = (const float*)d_in[4];
  const float* k_w    = (const float*)d_in[5];
  const float* v_w    = (const float*)d_in[6];
  const float* g_w    = (const float*)d_in[7];
  const float* g_b    = (const float*)d_in[8];
  const float* o_w    = (const float*)d_in[9];
  const float* o_b    = (const float*)d_in[10];
  float* out = (float*)d_out;

  char* base = (char*)d_ws;
  size_t off = 0;
  auto carve = [&](size_t bytes) -> char* {
    char* r = base + off;
    off = (off + bytes + 255) & ~(size_t)255;
    return r;
  };
  _Float16* Xq   = (_Float16*)carve((size_t)BS * DD * 2);
  _Float16* Xm   = (_Float16*)carve((size_t)BS * DD * 2);
  _Float16* Wq   = (_Float16*)carve((size_t)DD * DD * 2);
  _Float16* Wk   = (_Float16*)carve((size_t)DD * DD * 2);
  _Float16* Wv   = (_Float16*)carve((size_t)DD * DD * 2);
  _Float16* Wg   = (_Float16*)carve((size_t)DD * DD * 2);
  _Float16* Wo   = (_Float16*)carve((size_t)DD * DD * 2);
  _Float16* Qb   = (_Float16*)carve((size_t)BB * HH * SP * CC * 2);
  _Float16* Kb   = (_Float16*)carve((size_t)BB * HH * SP * CC * 2);
  _Float16* Vt   = (_Float16*)carve((size_t)BB * HH * SP * CC * 2);
  _Float16* gate = (_Float16*)carve((size_t)BS * DD * 2);
  _Float16* G2   = (_Float16*)carve((size_t)BS * DD * 2);

  const int n = BS * DD;  // 12,517,376
  k_cvt<<<dim3((n + 255) / 256), dim3(256), 0, stream>>>(q_data, Xq, n);
  k_cvt<<<dim3((n + 255) / 256), dim3(256), 0, stream>>>(m_data, Xm, n);
  k_cvt_tr256<<<dim3(256), dim3(256), 0, stream>>>(q_w, Wq);
  k_cvt_tr256<<<dim3(256), dim3(256), 0, stream>>>(k_w, Wk);
  k_cvt_tr256<<<dim3(256), dim3(256), 0, stream>>>(v_w, Wv);
  k_cvt_tr256<<<dim3(256), dim3(256), 0, stream>>>(g_w, Wg);
  k_cvt_tr256<<<dim3(256), dim3(256), 0, stream>>>(o_w, Wo);
  k_zero_pads<<<dim3(256), dim3(256), 0, stream>>>(Qb, Kb, Vt);
  k_proj<<<dim3(RT, 4), dim3(128), 0, stream>>>(Xq, Xm, Wq, Wk, Wv, Wg, g_b,
                                                Qb, Kb, Vt, gate);
  k_attn<<<dim3(BB * HH * 48 / 4), dim3(128), 0, stream>>>(Qb, Kb, Vt, bias,
                                                           nb, gate, G2);
  k_outproj<<<dim3(RT), dim3(128), 0, stream>>>(G2, Wo, o_b, out);
}
